// Beltrami_19267223290707
// MI455X (gfx1250) — matrix operand order repaired and checked by reading, not compiled, not run on hardware
//
#include <hip/hip_runtime.h>
#include <hip/hip_bf16.h>

// ---------------------------------------------------------------------------
// Problem constants (reference: B=8, N=4096, C=256, K=32)
// ---------------------------------------------------------------------------
#define BB   8
#define NN   4096
#define CC   256
#define KK   32
#define ROWS (BB * NN)          // 32768
#define DD   (2 * CC)           // 512
#define CHUNK 512               // sim columns per LDS chunk
#define NCHUNK (NN / CHUNK)     // 8

typedef __attribute__((ext_vector_type(16))) __bf16    v16bf;
typedef __attribute__((ext_vector_type(8)))  float     v8f;
typedef __attribute__((ext_vector_type(8)))  unsigned  v8u;

static __device__ __forceinline__ unsigned pack2_bf16(float a, float b) {
    unsigned ua = __builtin_bit_cast(unsigned, a);
    unsigned ub = __builtin_bit_cast(unsigned, b);
    ua = (ua + 0x7FFFu + ((ua >> 16) & 1u)) >> 16;   // round-to-nearest-even
    ub = (ub + 0x7FFFu + ((ub >> 16) & 1u)) >> 16;
    return (ub << 16) | (ua & 0xFFFFu);
}

// order-preserving float -> unsigned transform
static __device__ __forceinline__ unsigned fmono(float f) {
    unsigned u = __builtin_bit_cast(unsigned, f);
    return u ^ (unsigned)(((int)u >> 31) | 0x80000000);
}
static __device__ __forceinline__ float funmono(unsigned m) {
    unsigned u = (m & 0x80000000u) ? (m ^ 0x80000000u) : ~m;
    return __builtin_bit_cast(float, u);
}

// ---------------------------------------------------------------------------
// K0: convert f32 pairs -> packed bf16 (used for both W and x, one pass each)
// ---------------------------------------------------------------------------
__global__ void k_cvt_bf16(const float* __restrict__ src, unsigned* __restrict__ dst, int npair) {
    int i = blockIdx.x * blockDim.x + threadIdx.x;
    if (i < npair) {
        float2 v = *(const float2*)(src + 2 * (size_t)i);
        dst[i] = pack2_bf16(v.x, v.y);
    }
}

// ---------------------------------------------------------------------------
// K1: feat_pos = x @ W^T + bias   (32768x256 * 256x512) via bf16 WMMA, f32 out
// One wave computes a 16x64 output strip (4 column tiles) with A-fragments
// resident: 32 WMMAs per A-load.
// ---------------------------------------------------------------------------
__global__ __launch_bounds__(128) void k_gemm_featpos(
        const unsigned* __restrict__ xbf,     // (32768,256) bf16 -> 32768x128 uints
        const unsigned* __restrict__ Wbf,     // (512,256)   bf16 -> 512x128 uints
        const float*    __restrict__ bias,    // (512)
        float*          __restrict__ featpos) // (32768,512)
{
    const int lane  = threadIdx.x & 31;
    const int gwave = blockIdx.x * 4 + (threadIdx.x >> 5);   // 16384 waves
    const int mt = gwave >> 3;            // 2048 row tiles
    const int n0 = (gwave & 7) * 64;      // 8 column groups of 64
    const int m0 = mt * 16;
    const int lh = lane >> 4;
    const int ll = lane & 15;

    const unsigned* arow = xbf + (size_t)(m0 + ll) * (CC / 2);
    v8u afr[8];
#pragma unroll
    for (int kk = 0; kk < 8; ++kk) {
        const int ku = kk * 16 + lh * 4;              // uint offset of K = kk*32 + lh*8
        uint4 t0 = *(const uint4*)(arow + ku);
        uint4 t1 = *(const uint4*)(arow + ku + 8);    // K + 16
        afr[kk] = (v8u){ t0.x, t0.y, t0.z, t0.w, t1.x, t1.y, t1.z, t1.w };
    }

#pragma unroll
    for (int c = 0; c < 4; ++c) {
        const int nc = n0 + c * 16;
        const unsigned* wrow = Wbf + (size_t)(nc + ll) * (CC / 2);
        v8f acc = {};
#pragma unroll
        for (int kk = 0; kk < 8; ++kk) {
            const int bu = kk * 16 + lh * 4;
            uint4 b0 = *(const uint4*)(wrow + bu);
            uint4 b1 = *(const uint4*)(wrow + bu + 8);
            v8u bu8 = { b0.x, b0.y, b0.z, b0.w, b1.x, b1.y, b1.z, b1.w };
            acc = __builtin_amdgcn_wmma_f32_16x16x32_bf16(
                    false, __builtin_bit_cast(v16bf, afr[kk]),
                    false, __builtin_bit_cast(v16bf, bu8),
                    (short)0, acc, false, false);
        }
        const float bcol = bias[nc + ll];
        float* outc = featpos + (size_t)(m0 + lh * 8) * DD + (nc + ll);
#pragma unroll
        for (int v = 0; v < 8; ++v) outc[(size_t)v * DD] = acc[v] + bcol;
    }
}

// ---------------------------------------------------------------------------
// K2: normalize pos rows (featpos cols 256..511) -> packed bf16 (32768x256)
// ---------------------------------------------------------------------------
__global__ __launch_bounds__(128) void k_norm_pos(
        const float* __restrict__ featpos, unsigned* __restrict__ posbf)
{
    const int lane = threadIdx.x & 31;
    const int row  = blockIdx.x * 4 + (threadIdx.x >> 5);
    const float* p = featpos + (size_t)row * DD + CC + lane * 8;
    float4 p0 = *(const float4*)p;
    float4 p1 = *(const float4*)(p + 4);
    float ss = p0.x*p0.x + p0.y*p0.y + p0.z*p0.z + p0.w*p0.w
             + p1.x*p1.x + p1.y*p1.y + p1.z*p1.z + p1.w*p1.w;
#pragma unroll
    for (int off = 16; off; off >>= 1) ss += __shfl_xor(ss, off, 32);
    const float inv = 1.0f / fmaxf(sqrtf(ss), 1e-12f);
    uint4 o;
    o.x = pack2_bf16(p0.x * inv, p0.y * inv);
    o.y = pack2_bf16(p0.z * inv, p0.w * inv);
    o.z = pack2_bf16(p1.x * inv, p1.y * inv);
    o.w = pack2_bf16(p1.z * inv, p1.w * inv);
    *(uint4*)(posbf + (size_t)row * (CC / 2) + lane * 4) = o;
}

// ---------------------------------------------------------------------------
// K3: fused sim = pos @ pos^T (bf16 WMMA) + streaming top-32 via packed-key
//     replacement selection + softmax + gather.  WG = 16 rows of one batch.
// Key: [31:12] = monotonic f32 bits (truncated), [11:0] = 4095 - column.
// Larger key == larger value, ties -> smaller column (matches top_k order).
// Running top-32 of a row = one key per lane (kept[rr]).
// ---------------------------------------------------------------------------
__global__ __launch_bounds__(128) void k_topk_attn(
        const float*    __restrict__ featpos,  // (32768,512); feat = cols 0..255
        const unsigned* __restrict__ posbf,    // (32768,128) uints
        float*          __restrict__ out)      // (32768,256)
{
    __shared__ float simbuf[16][CHUNK];   // 32 KB
    __shared__ float attn_s[16][KK];
    __shared__ int   id_s[16][KK];

    const int lane = threadIdx.x & 31;
    const int wave = threadIdx.x >> 5;
    const int b    = blockIdx.x >> 8;          // 256 row-tiles per batch
    const int m0   = (blockIdx.x & 255) * 16;
    const int lh   = lane >> 4;
    const int ll   = lane & 15;

    // Load A fragments (this tile's 16 pos rows) once, keep resident.
    const unsigned* arow = posbf + (size_t)(b * NN + m0 + ll) * (CC / 2);
    v8u afr[8];
#pragma unroll
    for (int kk = 0; kk < 8; ++kk) {
        const int ku = kk * 16 + lh * 4;
        uint4 t0 = *(const uint4*)(arow + ku);
        uint4 t1 = *(const uint4*)(arow + ku + 8);
        afr[kk] = (v8u){ t0.x, t0.y, t0.z, t0.w, t1.x, t1.y, t1.z, t1.w };
    }

    unsigned kept[4] = { 0u, 0u, 0u, 0u };   // running top-32 keys, 1/lane/row

    for (int ch = 0; ch < NCHUNK; ++ch) {
        const int C0 = ch * CHUNK;
        // ---- sim tiles for this chunk (waves split the 32 column tiles) ----
        for (int t = wave; t < CHUNK / 16; t += 4) {
            const unsigned* brow = posbf + (size_t)(b * NN + C0 + t * 16 + ll) * (CC / 2);
            if (t + 4 < CHUNK / 16)
                __builtin_prefetch(brow + 4 * 16 * (CC / 2), 0, 3);
            v8f acc = {};
#pragma unroll
            for (int kk = 0; kk < 8; ++kk) {
                const int bu = kk * 16 + lh * 4;
                uint4 b0 = *(const uint4*)(brow + bu);
                uint4 b1 = *(const uint4*)(brow + bu + 8);
                v8u bu8 = { b0.x, b0.y, b0.z, b0.w, b1.x, b1.y, b1.z, b1.w };
                acc = __builtin_amdgcn_wmma_f32_16x16x32_bf16(
                        false, __builtin_bit_cast(v16bf, afr[kk]),
                        false, __builtin_bit_cast(v16bf, bu8),
                        (short)0, acc, false, false);
            }
#pragma unroll
            for (int v = 0; v < 8; ++v) simbuf[lh * 8 + v][t * 16 + ll] = acc[v];
        }
        __syncthreads();

        // ---- per-row replacement selection (register-resident, u32 max) ----
#pragma unroll
        for (int rr = 0; rr < 4; ++rr) {
            const int r = wave * 4 + rr;
            unsigned rv[16];
#pragma unroll
            for (int s = 0; s < 16; ++s) {
                const int col = C0 + lane + 32 * s;
                rv[s] = (fmono(simbuf[r][lane + 32 * s]) & 0xFFFFF000u)
                        | (unsigned)(4095 - col);
            }
            unsigned lmax = rv[0];
#pragma unroll
            for (int s = 1; s < 16; ++s) lmax = lmax > rv[s] ? lmax : rv[s];

            unsigned k0 = kept[rr];
            for (int guard = 0; guard < CHUNK; ++guard) {
                // wave max of candidates
                unsigned wmax = lmax;
#pragma unroll
                for (int off = 16; off; off >>= 1) {
                    unsigned o = __shfl_xor(wmax, off, 32);
                    wmax = wmax > o ? wmax : o;
                }
                // wave min of kept (+ owning lane)
                unsigned kmin = k0; int kml = lane;
#pragma unroll
                for (int off = 16; off; off >>= 1) {
                    unsigned ov = __shfl_xor(kmin, off, 32);
                    int      ol = __shfl_xor(kml,  off, 32);
                    if (ov < kmin || (ov == kmin && ol < kml)) { kmin = ov; kml = ol; }
                }
                if (wmax <= kmin) break;          // uniform across wave
                if (lane == kml) k0 = wmax;       // displace current minimum
                if (lmax == wmax) {               // winning lane repairs itself
                    unsigned nm = 0u;
#pragma unroll
                    for (int s = 0; s < 16; ++s) {
                        rv[s] = (rv[s] == wmax) ? 0u : rv[s];
                        nm = nm > rv[s] ? nm : rv[s];
                    }
                    lmax = nm;
                }
            }
            kept[rr] = k0;
        }
        __syncthreads();   // simbuf reused by next chunk
    }

    // ---- softmax over kept keys (one per lane), stash attn + ids in LDS ----
#pragma unroll
    for (int rr = 0; rr < 4; ++rr) {
        const int r = wave * 4 + rr;
        const unsigned key = kept[rr];
        const float v = funmono(key & 0xFFFFF000u);
        float mx = v;
#pragma unroll
        for (int off = 16; off; off >>= 1) mx = fmaxf(mx, __shfl_xor(mx, off, 32));
        float e = __expf(v - mx);
        float s = e;
#pragma unroll
        for (int off = 16; off; off >>= 1) s += __shfl_xor(s, off, 32);
        attn_s[r][lane] = e / s;
        id_s[r][lane]   = 4095 - (int)(key & 0xFFFu);
    }
    asm volatile("s_wait_dscnt 0" ::: "memory");

    // ---- gather: out[row,:] = sum_k attn[k] * feat[id_k,:]; lane owns 8 ch --
#pragma unroll
    for (int rr = 0; rr < 4; ++rr) {
        const int r = wave * 4 + rr;
        float a0 = 0.f, a1 = 0.f, a2 = 0.f, a3 = 0.f;
        float a4 = 0.f, a5 = 0.f, a6 = 0.f, a7 = 0.f;
        for (int k = 0; k < KK; ++k) {
            const float w  = attn_s[r][k];
            const int   id = id_s[r][k];
            const float* f = featpos + (size_t)(b * NN + id) * DD + lane * 8;
            float4 f0 = *(const float4*)f;
            float4 f1 = *(const float4*)(f + 4);
            a0 += w * f0.x; a1 += w * f0.y; a2 += w * f0.z; a3 += w * f0.w;
            a4 += w * f1.x; a5 += w * f1.y; a6 += w * f1.z; a7 += w * f1.w;
        }
        float* o = out + (size_t)(b * NN + m0 + r) * CC + lane * 8;
        *(float4*)o       = (float4){ a0, a1, a2, a3 };
        *(float4*)(o + 4) = (float4){ a4, a5, a6, a7 };
    }
}

// ---------------------------------------------------------------------------
// Host launcher
// ---------------------------------------------------------------------------
extern "C" void kernel_launch(void* const* d_in, const int* in_sizes, int n_in,
                              void* d_out, int out_size, void* d_ws, size_t ws_size,
                              hipStream_t stream) {
    const float* x    = (const float*)d_in[0];   // (8,4096,256)
    const float* W    = (const float*)d_in[1];   // (512,256)
    const float* bias = (const float*)d_in[2];   // (512)
    // d_in[3] is k (==32), compile-time constant here.

    char* ws = (char*)d_ws;
    size_t off = 0;
    unsigned* Wbf     = (unsigned*)(ws + off); off += (size_t)DD * CC / 2 * 4;         // 256 KB
    float*    featpos = (float*)(ws + off);    off += (size_t)ROWS * DD * 4;           // 64 MB
    unsigned* posbf   = (unsigned*)(ws + off); off += (size_t)ROWS * CC / 2 * 4;       // 16 MB
    unsigned* xbf     = (unsigned*)(ws + off);                                         // 16 MB

    const int wpairs = DD * CC / 2;        // 65536
    const int xpairs = ROWS * CC / 2;      // 4194304
    k_cvt_bf16     <<<(wpairs + 255) / 256, 256, 0, stream>>>(W, Wbf, wpairs);
    k_cvt_bf16     <<<(xpairs + 255) / 256, 256, 0, stream>>>(x, xbf, xpairs);
    k_gemm_featpos <<<4096,     128, 0, stream>>>(xbf, Wbf, bias, featpos);
    k_norm_pos     <<<ROWS / 4, 128, 0, stream>>>(featpos, posbf);
    k_topk_attn    <<<BB * (NN / 16), 128, 0, stream>>>(featpos, posbf, (float*)d_out);
}
